// MatchLSTM_6219112645470
// MI455X (gfx1250) — matrix operand order repaired and checked by reading, not compiled, not run on hardware
//
#include <hip/hip_runtime.h>
#include <hip/hip_bf16.h>

// ---------------------------------------------------------------------------
// Match-LSTM forward for MI455X (gfx1250, wave32, WMMA).
// All GEMMs use v_wmma_f32_16x16x32_bf16. Recurrences run as persistent
// single-workgroup kernels with register-resident gate accumulators.
// ---------------------------------------------------------------------------

#define DEVI __device__ __forceinline__

typedef __bf16 bf16_t;
typedef __attribute__((ext_vector_type(16))) __bf16 v16bf;
typedef __attribute__((ext_vector_type(8)))  float  v8f;

#define PP  400      // passage length
#define QQ  40       // question length
#define BB  32       // batch
#define HH  256      // hidden
#define EE  300      // embed dim
#define EP  320      // embed dim padded to K multiple of 32
#define G4  1024     // 4*H
#define H2  512      // 2*H

DEVI bf16_t f2bf(float f) {
  union { float f; unsigned u; } v; v.f = f;
  unsigned r = v.u + 0x7FFFu + ((v.u >> 16) & 1u);   // round-to-nearest-even
  union { unsigned short u; bf16_t b; } o; o.u = (unsigned short)(r >> 16);
  return o.b;
}
DEVI bf16_t bf0() { union { unsigned short u; bf16_t b; } o; o.u = 0; return o.b; }
DEVI float  sigm(float x) { return 1.f / (1.f + __expf(-x)); }
DEVI v8f    vzero8() { v8f z; for (int i = 0; i < 8; ++i) z[i] = 0.f; return z; }

// A fragment (16x32 bf16 tile of row-major A). Lanes 0-15: rows 0-15, K 0-15;
// lanes 16-31: same rows, K 16-31 (matches load_b_frag k order).
DEVI v16bf load_a_frag(const bf16_t* A, int lda, int row0, int k0, int lane) {
  int n16 = lane & 15, half = lane >> 4;
  const bf16_t* p = A + (size_t)(row0 + n16) * lda + k0 + half * 16;
  v16bf f;
#pragma unroll
  for (int j = 0; j < 8; ++j) f[j] = p[j];
#pragma unroll
  for (int j = 0; j < 8; ++j) f[8 + j] = p[8 + j];
  return f;
}
// B fragment for C = A @ W^T: B_eff[k][n] = W[n0+n][k0+k], W row-major [N][K].
// Lane n = lane&15, K offset (lane>>4)*16, 16 consecutive k per lane.
DEVI v16bf load_b_frag(const bf16_t* W, int ldw, int n0, int k0, int lane) {
  int n16 = lane & 15, koff = (lane >> 4) * 16;
  const bf16_t* p = W + (size_t)(n0 + n16) * ldw + k0 + koff;
  v16bf f;
#pragma unroll
  for (int j = 0; j < 16; ++j) f[j] = p[j];
  return f;
}
DEVI v8f wmma_bf16(v16bf a, v16bf b, v8f c) {
  return __builtin_amdgcn_wmma_f32_16x16x32_bf16(false, a, false, b, (short)0, c,
                                                 false, false);
}

// ---------------------------------------------------------------------------
// Format conversion / gather kernels
// ---------------------------------------------------------------------------
__global__ void cvt_kernel(bf16_t* dst, const float* src, int n) {
  int i = blockIdx.x * 256 + threadIdx.x;
  if (i < n) dst[i] = f2bf(src[i]);
}
__global__ void cvt_pad_kernel(bf16_t* dst, const float* src, int rows, int cols,
                               int ldd) {
  int i = blockIdx.x * 256 + threadIdx.x;
  if (i >= rows * ldd) return;
  int r = i / ldd, c = i % ldd;
  dst[i] = (c < cols) ? f2bf(src[(size_t)r * cols + c]) : bf0();
}
// dst[c][r] = src[r][c]  (for X@W with un-transposed W: store W^T)
__global__ void cvt_T_kernel(bf16_t* dst, const float* src, int rows, int cols) {
  int i = blockIdx.x * 256 + threadIdx.x;
  if (i >= rows * cols) return;
  int r = i / cols, c = i % cols;
  dst[(size_t)c * rows + r] = f2bf(src[i]);
}
__global__ void gather_kernel(bf16_t* X, const int* ids, const float* embed) {
  int row = blockIdx.x;                       // t*B + b
  int id = ids[row];
  const float* e = embed + (size_t)id * EE;
  for (int k = threadIdx.x; k < EP; k += blockDim.x)
    X[(size_t)row * EP + k] = (k < EE) ? f2bf(e[k]) : bf0();
}

// ---------------------------------------------------------------------------
// Generic C[M,N] = A[M,K] @ W[N,K]^T + bias.  8 waves/block, one 16x16 tile
// per wave.  grid = (M/16, ceil(N/128)).
// ---------------------------------------------------------------------------
__global__ __launch_bounds__(256) void wmma_gemm_kernel(
    const bf16_t* __restrict__ A, int lda, const bf16_t* __restrict__ W, int ldw,
    const float* __restrict__ bias, float* __restrict__ C, int ldc, int N, int K) {
  int lane = threadIdx.x & 31, wave = threadIdx.x >> 5;
  int row0 = blockIdx.x * 16;
  int col0 = (blockIdx.y * 8 + wave) * 16;
  if (col0 >= N) return;                       // wave-uniform exit
  v8f acc = vzero8();
  for (int k0 = 0; k0 < K; k0 += 32) {
    v16bf a = load_a_frag(A, lda, row0, k0, lane);
    v16bf b = load_b_frag(W, ldw, col0, k0, lane);
    acc = wmma_bf16(a, b, acc);
  }
  int n16 = lane & 15, half = lane >> 4;
  float bv = bias ? bias[col0 + n16] : 0.f;
  float* cp = C + (size_t)(row0 + half * 8) * ldc + col0 + n16;
#pragma unroll
  for (int i = 0; i < 8; ++i) cp[(size_t)i * ldc] = acc[i] + bv;
}

// ---------------------------------------------------------------------------
// Preprocess LSTM (passage blockIdx 0, question blockIdx 1).  One WG, 8 waves.
// G holds x@Wih^T + b precomputed.  Module quirk: h := tanh(c)*mask (o-gate
// unused), so only 3 of 4 gate tile groups are computed.
// Wave w owns gate column tiles {w, w+8} for all 3 gates and both M tiles:
// gate accumulators and cell state c live entirely in registers.
// ---------------------------------------------------------------------------
__global__ __launch_bounds__(256) void preproc_kernel(
    const float* __restrict__ Gp, const float* __restrict__ Gq,
    const bf16_t* __restrict__ Whh, const int* __restrict__ plens,
    const int* __restrict__ qlens, float* __restrict__ Hp,
    bf16_t* __restrict__ Hpb, float* __restrict__ Hq, bf16_t* __restrict__ Hqb) {
  const bool isQ = (blockIdx.x == 1);
  const float* G = isQ ? Gq : Gp;
  const int* lens = isQ ? qlens : plens;
  float* Ho = isQ ? Hq : Hp;
  bf16_t* Hob = isQ ? Hqb : Hpb;
  const int T = isQ ? QQ : PP;

  __shared__ bf16_t h_s[BB * HH];              // 16 KB
  int tid = threadIdx.x, lane = tid & 31, wave = tid >> 5;
  int n16 = lane & 15, half = lane >> 4;
  for (int i = tid; i < BB * HH; i += 256) h_s[i] = bf0();
  float c_reg[2][2][8];
  for (int a = 0; a < 2; ++a)
    for (int b = 0; b < 2; ++b)
      for (int i = 0; i < 8; ++i) c_reg[a][b][i] = 0.f;
  __syncthreads();

  for (int t = 0; t < T; ++t) {
    v8f acc[2][2][3];
#pragma unroll
    for (int mt = 0; mt < 2; ++mt)
#pragma unroll
      for (int u2 = 0; u2 < 2; ++u2)
#pragma unroll
        for (int g = 0; g < 3; ++g) acc[mt][u2][g] = vzero8();
    for (int k0 = 0; k0 < HH; k0 += 32) {
      v16bf a0 = load_a_frag(h_s, HH, 0, k0, lane);
      v16bf a1 = load_a_frag(h_s, HH, 16, k0, lane);
#pragma unroll
      for (int u2 = 0; u2 < 2; ++u2) {
        int nt = wave + 8 * u2;
#pragma unroll
        for (int g = 0; g < 3; ++g) {
          v16bf bfr = load_b_frag(Whh, HH, g * HH + nt * 16, k0, lane);
          acc[0][u2][g] = wmma_bf16(a0, bfr, acc[0][u2][g]);
          acc[1][u2][g] = wmma_bf16(a1, bfr, acc[1][u2][g]);
        }
      }
    }
    __syncthreads();                           // all waves done reading h_s
#pragma unroll
    for (int mt = 0; mt < 2; ++mt)
#pragma unroll
      for (int u2 = 0; u2 < 2; ++u2) {
        int u = (wave + 8 * u2) * 16 + n16;
#pragma unroll
        for (int i = 0; i < 8; ++i) {
          int b = mt * 16 + half * 8 + i;
          const float* grow = G + ((size_t)t * BB + b) * G4;
          float gi = acc[mt][u2][0][i] + grow[u];
          float gf = acc[mt][u2][1][i] + grow[HH + u];
          float gg = acc[mt][u2][2][i] + grow[2 * HH + u];
          float c2 = sigm(gf) * c_reg[mt][u2][i] + sigm(gi) * tanhf(gg);
          float m = (t < lens[b]) ? 1.f : 0.f;
          float h2 = tanhf(c2) * m;            // quirk: h := tanh(c)*mask
          c_reg[mt][u2][i] = c2 * m;
          size_t orow = ((size_t)t * BB + b) * HH + u;
          Ho[orow] = h2;
          bf16_t hb = f2bf(h2);
          Hob[orow] = hb;
          h_s[b * HH + u] = hb;
        }
      }
    __syncthreads();
  }
}

// ---------------------------------------------------------------------------
// Match-LSTM with attention over the question.  grid=2 (dir 0 fwd, 1 bwd),
// each direction on its own WGP.  LDS ~92 KB.
// ---------------------------------------------------------------------------
__global__ __launch_bounds__(256) void match_kernel(
    const float* __restrict__ attQ, const float* __restrict__ Hq,
    const float* __restrict__ Hp, const bf16_t* __restrict__ Hpb,
    const int* __restrict__ plens, const bf16_t* __restrict__ WpT,
    const bf16_t* __restrict__ WhT, const float* __restrict__ bp,
    const float* __restrict__ w_alpha, const float* __restrict__ b_alpha,
    const bf16_t* __restrict__ mWih, const bf16_t* __restrict__ mWhh,
    const float* __restrict__ m_b, float* __restrict__ Hr,
    bf16_t* __restrict__ Hrb) {
  const int dir = blockIdx.x;
  __shared__ bf16_t h_s[BB * HH];              // 16 KB  match hidden (bf16)
  __shared__ bf16_t zt_s[BB * H2];             // 32 KB  LSTM input [hp, wq]
  __shared__ float  pv_s[BB * HH];             // 32 KB  hp@Wp + h@Wh + bp
  __shared__ float  s_s[QQ * BB];              //  5 KB  attention logits
  __shared__ float  al_s[QQ * BB];             //  5 KB  softmax alpha
  __shared__ float  wal_s[HH];                 //  1 KB  w_alpha copy
  int tid = threadIdx.x, lane = tid & 31, wave = tid >> 5;
  int n16 = lane & 15, half = lane >> 4;
  for (int i = tid; i < BB * HH; i += 256) h_s[i] = bf0();
  for (int i = tid; i < HH; i += 256) wal_s[i] = w_alpha[i];
  float c_reg[2][2][8];
  for (int a = 0; a < 2; ++a)
    for (int b = 0; b < 2; ++b)
      for (int i = 0; i < 8; ++i) c_reg[a][b][i] = 0.f;
  float balpha = b_alpha[0];
  __syncthreads();

  for (int t = 0; t < PP; ++t) {
    int p = dir ? (PP - 1 - t) : t;
    // ---- pvec = Hp[p]@Wp^T(+T) + h@Wh^T(+T) + bp ---------------------------
#pragma unroll
    for (int u2 = 0; u2 < 2; ++u2) {
      int nt = wave + 8 * u2;
#pragma unroll
      for (int mt = 0; mt < 2; ++mt) {
        v8f acc = vzero8();
        for (int k0 = 0; k0 < HH; k0 += 32) {
          v16bf a = load_a_frag(Hpb, HH, p * BB + mt * 16, k0, lane);
          v16bf b = load_b_frag(WpT, HH, nt * 16, k0, lane);
          acc = wmma_bf16(a, b, acc);
        }
        for (int k0 = 0; k0 < HH; k0 += 32) {
          v16bf a = load_a_frag(h_s, HH, mt * 16, k0, lane);
          v16bf b = load_b_frag(WhT, HH, nt * 16, k0, lane);
          acc = wmma_bf16(a, b, acc);
        }
        float bb = bp[nt * 16 + n16];
        int r0 = mt * 16 + half * 8, cc = nt * 16 + n16;
#pragma unroll
        for (int i = 0; i < 8; ++i) pv_s[(r0 + i) * HH + cc] = acc[i] + bb;
      }
    }
    __syncthreads();
    // ---- attention logits: s[q,b] = sum_u tanh(attQ+pvec)*w_alpha + b -----
    for (int idx = tid; idx < QQ * BB; idx += 256) {
      int q = idx >> 5, b = idx & 31;
      const float* aq = attQ + ((size_t)q * BB + b) * HH;
      const float* pv = pv_s + b * HH;
      float s = balpha;
      for (int u = 0; u < HH; ++u) s += tanhf(aq[u] + pv[u]) * wal_s[u];
      s_s[idx] = s;
    }
    __syncthreads();
    // ---- softmax over q (lane b) ------------------------------------------
    if (tid < BB) {
      int b = tid;
      float mx = -3.0e38f;
      for (int q = 0; q < QQ; ++q) mx = fmaxf(mx, s_s[q * BB + b]);
      float sum = 0.f;
      for (int q = 0; q < QQ; ++q) {
        float e = __expf(s_s[q * BB + b] - mx);
        al_s[q * BB + b] = e;
        sum += e;
      }
      float inv = 1.f / sum;
      for (int q = 0; q < QQ; ++q) al_s[q * BB + b] *= inv;
    }
    __syncthreads();
    // ---- wq = sum_q alpha*Hq;  zt = [hp, wq]*mask (bf16 in LDS) -----------
    for (int idx = tid; idx < BB * HH; idx += 256) {
      int b = idx >> 8, u = idx & 255;
      float wq = 0.f;
      for (int q = 0; q < QQ; ++q)
        wq += al_s[q * BB + b] * Hq[((size_t)q * BB + b) * HH + u];
      float m = (p < plens[b]) ? 1.f : 0.f;
      float hp = Hp[((size_t)p * BB + b) * HH + u];
      zt_s[b * H2 + u] = f2bf(hp * m);
      zt_s[b * H2 + HH + u] = f2bf(wq * m);
    }
    __syncthreads();
    // ---- gates = zt@mWih^T + h@mWhh^T + m_b (register-resident accs) ------
    v8f acc[2][2][4];
#pragma unroll
    for (int mt = 0; mt < 2; ++mt)
#pragma unroll
      for (int u2 = 0; u2 < 2; ++u2)
#pragma unroll
        for (int g = 0; g < 4; ++g) acc[mt][u2][g] = vzero8();
    for (int k0 = 0; k0 < H2; k0 += 32) {
      v16bf a0 = load_a_frag(zt_s, H2, 0, k0, lane);
      v16bf a1 = load_a_frag(zt_s, H2, 16, k0, lane);
#pragma unroll
      for (int u2 = 0; u2 < 2; ++u2) {
        int nt = wave + 8 * u2;
#pragma unroll
        for (int g = 0; g < 4; ++g) {
          v16bf bfr = load_b_frag(mWih, H2, g * HH + nt * 16, k0, lane);
          acc[0][u2][g] = wmma_bf16(a0, bfr, acc[0][u2][g]);
          acc[1][u2][g] = wmma_bf16(a1, bfr, acc[1][u2][g]);
        }
      }
    }
    for (int k0 = 0; k0 < HH; k0 += 32) {
      v16bf a0 = load_a_frag(h_s, HH, 0, k0, lane);
      v16bf a1 = load_a_frag(h_s, HH, 16, k0, lane);
#pragma unroll
      for (int u2 = 0; u2 < 2; ++u2) {
        int nt = wave + 8 * u2;
#pragma unroll
        for (int g = 0; g < 4; ++g) {
          v16bf bfr = load_b_frag(mWhh, HH, g * HH + nt * 16, k0, lane);
          acc[0][u2][g] = wmma_bf16(a0, bfr, acc[0][u2][g]);
          acc[1][u2][g] = wmma_bf16(a1, bfr, acc[1][u2][g]);
        }
      }
    }
    __syncthreads();                           // done reading h_s / zt_s
    // ---- LSTM cell update (full, with o-gate) -----------------------------
#pragma unroll
    for (int mt = 0; mt < 2; ++mt)
#pragma unroll
      for (int u2 = 0; u2 < 2; ++u2) {
        int u = (wave + 8 * u2) * 16 + n16;
#pragma unroll
        for (int i = 0; i < 8; ++i) {
          int b = mt * 16 + half * 8 + i;
          float gi = acc[mt][u2][0][i] + m_b[u];
          float gf = acc[mt][u2][1][i] + m_b[HH + u];
          float gg = acc[mt][u2][2][i] + m_b[2 * HH + u];
          float go = acc[mt][u2][3][i] + m_b[3 * HH + u];
          float c2 = sigm(gf) * c_reg[mt][u2][i] + sigm(gi) * tanhf(gg);
          float m = (p < plens[b]) ? 1.f : 0.f;
          float h2 = sigm(go) * tanhf(c2) * m;
          c_reg[mt][u2][i] = c2 * m;
          h_s[b * HH + u] = f2bf(h2);
          size_t row = ((size_t)p * BB + b) * H2 + (size_t)dir * HH + u;
          Hr[row] = h2;
          Hrb[row] = f2bf(h2);
        }
      }
    __syncthreads();
  }
}

// ---------------------------------------------------------------------------
// Pointer network pieces
// ---------------------------------------------------------------------------
__global__ void ptr_scores_kernel(const float* __restrict__ attM,
                                  const float* __restrict__ aW,
                                  const float* __restrict__ w_beta,
                                  const float* __restrict__ b_beta,
                                  const int* __restrict__ plens,
                                  float* __restrict__ scores) {
  int idx = blockIdx.x * 256 + threadIdx.x;    // p*32 + b
  if (idx >= PP * BB) return;
  int b = idx & 31, p = idx >> 5;
  const float* am = attM + (size_t)idx * HH;
  const float* aw = aW + b * HH;
  float s = b_beta[0];
  for (int u = 0; u < HH; ++u) s += tanhf(am[u] + aw[u]) * w_beta[u];
  scores[idx] = (p < plens[b]) ? s : -1e30f;
}

__global__ void ptr_softmax_kernel(const float* __restrict__ scores,
                                   const int* __restrict__ plens,
                                   float* __restrict__ beta,
                                   float* __restrict__ out) {   // out: [B][P]
  int b = threadIdx.x;
  if (b >= BB) return;
  float mx = -3.0e38f;
  for (int p = 0; p < PP; ++p) mx = fmaxf(mx, scores[p * BB + b]);
  float sum = 0.f;
  for (int p = 0; p < PP; ++p) {
    float e = __expf(scores[p * BB + b] - mx);
    beta[p * BB + b] = e;
    sum += e;
  }
  float inv = 1.f / sum;
  for (int p = 0; p < PP; ++p) {
    float v = beta[p * BB + b] * inv;
    if (p >= plens[b]) v = 0.f;                // beta *= valid
    beta[p * BB + b] = v;
    out[(size_t)b * PP + p] = v;
  }
}

__global__ void ptr_whr_kernel(const float* __restrict__ beta,
                               const float* __restrict__ Hr,
                               bf16_t* __restrict__ wHrb) {
  int idx = blockIdx.x * 256 + threadIdx.x;    // b*512 + j
  if (idx >= BB * H2) return;
  int b = idx >> 9, j = idx & 511;
  float s = 0.f;
  for (int p = 0; p < PP; ++p)
    s += beta[p * BB + b] * Hr[((size_t)p * BB + b) * H2 + j];
  wHrb[idx] = f2bf(s);
}

// Pointer LSTM cell (initial c == 0, so the f-gate term vanishes).
__global__ __launch_bounds__(256) void ptr_cell_kernel(
    const bf16_t* __restrict__ wHrb, const bf16_t* __restrict__ hab,
    const bf16_t* __restrict__ pWih, const bf16_t* __restrict__ pWhh,
    const float* __restrict__ p_b, bf16_t* __restrict__ haOut) {
  int tid = threadIdx.x, lane = tid & 31, wave = tid >> 5;
  int n16 = lane & 15, half = lane >> 4;
  v8f acc[2][2][4];
#pragma unroll
  for (int mt = 0; mt < 2; ++mt)
#pragma unroll
    for (int u2 = 0; u2 < 2; ++u2)
#pragma unroll
      for (int g = 0; g < 4; ++g) acc[mt][u2][g] = vzero8();
  for (int k0 = 0; k0 < H2; k0 += 32) {
    v16bf a0 = load_a_frag(wHrb, H2, 0, k0, lane);
    v16bf a1 = load_a_frag(wHrb, H2, 16, k0, lane);
#pragma unroll
    for (int u2 = 0; u2 < 2; ++u2) {
      int nt = wave + 8 * u2;
#pragma unroll
      for (int g = 0; g < 4; ++g) {
        v16bf bfr = load_b_frag(pWih, H2, g * HH + nt * 16, k0, lane);
        acc[0][u2][g] = wmma_bf16(a0, bfr, acc[0][u2][g]);
        acc[1][u2][g] = wmma_bf16(a1, bfr, acc[1][u2][g]);
      }
    }
  }
  for (int k0 = 0; k0 < HH; k0 += 32) {
    v16bf a0 = load_a_frag(hab, HH, 0, k0, lane);
    v16bf a1 = load_a_frag(hab, HH, 16, k0, lane);
#pragma unroll
    for (int u2 = 0; u2 < 2; ++u2) {
      int nt = wave + 8 * u2;
#pragma unroll
      for (int g = 0; g < 4; ++g) {
        v16bf bfr = load_b_frag(pWhh, HH, g * HH + nt * 16, k0, lane);
        acc[0][u2][g] = wmma_bf16(a0, bfr, acc[0][u2][g]);
        acc[1][u2][g] = wmma_bf16(a1, bfr, acc[1][u2][g]);
      }
    }
  }
#pragma unroll
  for (int mt = 0; mt < 2; ++mt)
#pragma unroll
    for (int u2 = 0; u2 < 2; ++u2) {
      int u = (wave + 8 * u2) * 16 + n16;
#pragma unroll
      for (int i = 0; i < 8; ++i) {
        int b = mt * 16 + half * 8 + i;
        float gi = acc[mt][u2][0][i] + p_b[u];
        float gg = acc[mt][u2][2][i] + p_b[2 * HH + u];
        float go = acc[mt][u2][3][i] + p_b[3 * HH + u];
        float c2 = sigm(gi) * tanhf(gg);       // c0 == 0
        float h2 = sigm(go) * tanhf(c2);
        haOut[b * HH + u] = f2bf(h2);
      }
    }
}

// ---------------------------------------------------------------------------
// Host orchestration
// ---------------------------------------------------------------------------
extern "C" void kernel_launch(void* const* d_in, const int* in_sizes, int n_in,
                              void* d_out, int out_size, void* d_ws,
                              size_t ws_size, hipStream_t stream) {
  (void)in_sizes; (void)n_in; (void)out_size; (void)ws_size;
  const int*   passage_ids  = (const int*)d_in[0];
  const int*   question_ids = (const int*)d_in[1];
  const int*   plens        = (const int*)d_in[2];
  const int*   qlens        = (const int*)d_in[3];
  /* d_in[4] = answer: unused by forward */
  const float* embed   = (const float*)d_in[5];
  const float* pre_Wih = (const float*)d_in[6];
  const float* pre_Whh = (const float*)d_in[7];
  const float* pre_b   = (const float*)d_in[8];
  const float* Wq      = (const float*)d_in[9];
  const float* Wp      = (const float*)d_in[10];
  const float* bp      = (const float*)d_in[11];
  const float* Wh      = (const float*)d_in[12];
  const float* w_alpha = (const float*)d_in[13];
  const float* b_alpha = (const float*)d_in[14];
  const float* m_Wih   = (const float*)d_in[15];
  const float* m_Whh   = (const float*)d_in[16];
  const float* m_b     = (const float*)d_in[17];
  const float* Wm      = (const float*)d_in[18];
  const float* Wa      = (const float*)d_in[19];
  const float* ba      = (const float*)d_in[20];
  const float* w_beta  = (const float*)d_in[21];
  const float* b_beta  = (const float*)d_in[22];
  const float* p_Wih   = (const float*)d_in[23];
  const float* p_Whh   = (const float*)d_in[24];
  const float* p_b     = (const float*)d_in[25];
  float* out = (float*)d_out;

  char* base = (char*)d_ws;
  size_t off = 0;
  auto alloc = [&](size_t bytes) -> void* {
    void* p = base + off;
    off += (bytes + 255) & ~(size_t)255;
    return p;
  };
  bf16_t* Wih_bf  = (bf16_t*)alloc((size_t)G4 * EP * 2);
  bf16_t* Whh_bf  = (bf16_t*)alloc((size_t)G4 * HH * 2);
  bf16_t* WqT_bf  = (bf16_t*)alloc((size_t)HH * HH * 2);
  bf16_t* WpT_bf  = (bf16_t*)alloc((size_t)HH * HH * 2);
  bf16_t* WhT_bf  = (bf16_t*)alloc((size_t)HH * HH * 2);
  bf16_t* WaT_bf  = (bf16_t*)alloc((size_t)HH * HH * 2);
  bf16_t* WmT_bf  = (bf16_t*)alloc((size_t)HH * H2 * 2);
  bf16_t* mWih_bf = (bf16_t*)alloc((size_t)G4 * H2 * 2);
  bf16_t* mWhh_bf = (bf16_t*)alloc((size_t)G4 * HH * 2);
  bf16_t* pWih_bf = (bf16_t*)alloc((size_t)G4 * H2 * 2);
  bf16_t* pWhh_bf = (bf16_t*)alloc((size_t)G4 * HH * 2);
  bf16_t* Xp_bf   = (bf16_t*)alloc((size_t)PP * BB * EP * 2);
  bf16_t* Xq_bf   = (bf16_t*)alloc((size_t)QQ * BB * EP * 2);
  float*  Gp      = (float*) alloc((size_t)PP * BB * G4 * 4);
  float*  Gq      = (float*) alloc((size_t)QQ * BB * G4 * 4);
  float*  Hp_f    = (float*) alloc((size_t)PP * BB * HH * 4);
  bf16_t* Hp_bf   = (bf16_t*)alloc((size_t)PP * BB * HH * 2);
  float*  Hq_f    = (float*) alloc((size_t)QQ * BB * HH * 4);
  bf16_t* Hq_bf   = (bf16_t*)alloc((size_t)QQ * BB * HH * 2);
  float*  attQ    = (float*) alloc((size_t)QQ * BB * HH * 4);
  float*  Hr_f    = (float*) alloc((size_t)PP * BB * H2 * 4);
  bf16_t* Hr_bf   = (bf16_t*)alloc((size_t)PP * BB * H2 * 2);
  float*  attM    = (float*) alloc((size_t)PP * BB * HH * 4);
  float*  aW      = (float*) alloc((size_t)BB * HH * 4);
  bf16_t* ha0_bf  = (bf16_t*)alloc((size_t)BB * HH * 2);
  bf16_t* ha1_bf  = (bf16_t*)alloc((size_t)BB * HH * 2);
  float*  scoresb = (float*) alloc((size_t)PP * BB * 4);
  float*  betab   = (float*) alloc((size_t)PP * BB * 4);
  bf16_t* wHr_bf  = (bf16_t*)alloc((size_t)BB * H2 * 2);

  const int thr = 256;
  // -- weight conversions (bf16, W^T where math is X@W not X@W^T) ----------
  cvt_pad_kernel<<<(G4 * EP + 255) / 256, thr, 0, stream>>>(Wih_bf, pre_Wih, G4, EE, EP);
  cvt_kernel<<<(G4 * HH + 255) / 256, thr, 0, stream>>>(Whh_bf, pre_Whh, G4 * HH);
  cvt_kernel<<<(G4 * H2 + 255) / 256, thr, 0, stream>>>(mWih_bf, m_Wih, G4 * H2);
  cvt_kernel<<<(G4 * HH + 255) / 256, thr, 0, stream>>>(mWhh_bf, m_Whh, G4 * HH);
  cvt_kernel<<<(G4 * H2 + 255) / 256, thr, 0, stream>>>(pWih_bf, p_Wih, G4 * H2);
  cvt_kernel<<<(G4 * HH + 255) / 256, thr, 0, stream>>>(pWhh_bf, p_Whh, G4 * HH);
  cvt_T_kernel<<<(HH * HH + 255) / 256, thr, 0, stream>>>(WqT_bf, Wq, HH, HH);
  cvt_T_kernel<<<(HH * HH + 255) / 256, thr, 0, stream>>>(WpT_bf, Wp, HH, HH);
  cvt_T_kernel<<<(HH * HH + 255) / 256, thr, 0, stream>>>(WhT_bf, Wh, HH, HH);
  cvt_T_kernel<<<(HH * HH + 255) / 256, thr, 0, stream>>>(WaT_bf, Wa, HH, HH);
  cvt_T_kernel<<<(H2 * HH + 255) / 256, thr, 0, stream>>>(WmT_bf, Wm, H2, HH);
  // -- embedding gather (bf16, K padded to 320) ----------------------------
  gather_kernel<<<PP * BB, 64, 0, stream>>>(Xp_bf, passage_ids, embed);
  gather_kernel<<<QQ * BB, 64, 0, stream>>>(Xq_bf, question_ids, embed);
  // -- hoisted input projections: G = X @ Wih^T + b (fully parallel) -------
  wmma_gemm_kernel<<<dim3(PP * BB / 16, G4 / 128), thr, 0, stream>>>(
      Xp_bf, EP, Wih_bf, EP, pre_b, Gp, G4, G4, EP);
  wmma_gemm_kernel<<<dim3(QQ * BB / 16, G4 / 128), thr, 0, stream>>>(
      Xq_bf, EP, Wih_bf, EP, pre_b, Gq, G4, G4, EP);
  hipMemsetAsync(ha0_bf, 0, (size_t)BB * HH * 2, stream);
  // -- preprocess recurrences (passage + question, one WG each) ------------
  preproc_kernel<<<2, thr, 0, stream>>>(Gp, Gq, Whh_bf, plens, qlens, Hp_f, Hp_bf,
                                        Hq_f, Hq_bf);
  // -- attQ = Hq @ Wq ------------------------------------------------------
  wmma_gemm_kernel<<<dim3(QQ * BB / 16, HH / 128), thr, 0, stream>>>(
      Hq_bf, HH, WqT_bf, HH, nullptr, attQ, HH, HH, HH);
  // -- bidirectional match-LSTM (dir per block) ----------------------------
  match_kernel<<<2, thr, 0, stream>>>(attQ, Hq_f, Hp_f, Hp_bf, plens, WpT_bf,
                                      WhT_bf, bp, w_alpha, b_alpha, mWih_bf,
                                      mWhh_bf, m_b, Hr_f, Hr_bf);
  // -- attM = Hr @ Wm ------------------------------------------------------
  wmma_gemm_kernel<<<dim3(PP * BB / 16, HH / 128), thr, 0, stream>>>(
      Hr_bf, H2, WmT_bf, H2, nullptr, attM, HH, HH, H2);
  // -- pointer network (2 steps) -------------------------------------------
  for (int k = 0; k < 2; ++k) {
    const bf16_t* ha = k ? ha1_bf : ha0_bf;
    wmma_gemm_kernel<<<dim3(BB / 16, HH / 128), thr, 0, stream>>>(
        ha, HH, WaT_bf, HH, ba, aW, HH, HH, HH);
    ptr_scores_kernel<<<(PP * BB + 255) / 256, thr, 0, stream>>>(
        attM, aW, w_beta, b_beta, plens, scoresb);
    ptr_softmax_kernel<<<1, 32, 0, stream>>>(scoresb, plens, betab,
                                             out + (size_t)k * BB * PP);
    if (k == 0) {
      ptr_whr_kernel<<<(BB * H2 + 255) / 256, thr, 0, stream>>>(betab, Hr_f,
                                                                wHr_bf);
      ptr_cell_kernel<<<1, thr, 0, stream>>>(wHr_bf, ha0_bf, pWih_bf, pWhh_bf,
                                             p_b, ha1_bf);
    }
  }
}
